// PedestrianEncoder_29798483100408
// MI455X (gfx1250) — compile-verified, matrix operand order checked by reading, not run on hardware
//
#include <hip/hip_runtime.h>

// ---------------------------------------------------------------------------
// Problem constants (match reference setup_inputs)
// ---------------------------------------------------------------------------
constexpr int BB = 32;    // batch
constexpr int NN = 64;    // pedestrians
constexpr int TT = 12;    // time steps
constexpr int CC = 2;     // channels
constexpr int MM = 20;    // modes
constexpr int HH = 128;   // hidden
constexpr int G3 = 384;   // 3*H (gates)
constexpr int FF = 64;    // feature_dim
constexpr int NSEQ = BB * NN * MM;  // 40960 GRU sequences
constexpr int NBN  = BB * NN;       // 2048 (b,n) groups

typedef __attribute__((ext_vector_type(16))) __bf16 v16bf;
typedef __attribute__((ext_vector_type(8)))  __bf16 v8bf;
typedef __attribute__((ext_vector_type(8)))  float  v8f;
typedef __attribute__((ext_vector_type(4)))  float  v4f;

__device__ __forceinline__ v8f splat8(float x) {
  v8f r;
#pragma unroll
  for (int i = 0; i < 8; ++i) r[i] = x;
  return r;
}

// Hardware v_rcp_f32 (1 TRANS op) instead of the IEEE div expansion.
__device__ __forceinline__ float fast_rcp(float x) {
  return __builtin_amdgcn_rcpf(x);
}
// sigmoid(x) = 1/(1+e^-x): v_exp + v_rcp, co-executes with XDL WMMA.
__device__ __forceinline__ float fast_sigmoid(float x) {
  return fast_rcp(1.0f + __expf(-x));
}
// tanh(a) = 2/(1+e^-2a) - 1 (saturates correctly: exp->inf => rcp->0 => -1).
__device__ __forceinline__ float fast_tanh(float a) {
  return 2.0f * fast_rcp(1.0f + __expf(-2.0f * a)) - 1.0f;
}

// D = A*B + C, bf16 operands, f32 accum  -> v_wmma_f32_16x16x32_bf16
__device__ __forceinline__ v8f wmma_bf16(v16bf a, v16bf b, v8f c) {
  return __builtin_amdgcn_wmma_f32_16x16x32_bf16(false, a, false, b,
                                                 (short)0, c, false, false);
}

// Load a 16x32 bf16 A (or 32x16 B) fragment from LDS, row-major source.
// Per ISA 7.12.2: lane l (r = l&15, half = l>>4) holds K = k0+half*8+0..7
// and K = k0+16+half*8+0..7 for row/col r0+r.
__device__ __forceinline__ v16bf frag_from_lds(const __bf16* base, int ld,
                                               int r0, int k0, int lane) {
  int r  = r0 + (lane & 15);
  int kb = k0 + ((lane >> 4) << 3);
  const __bf16* p = base + r * ld + kb;
  union { v16bf v; v8bf h[2]; } u;
  u.h[0] = *(const v8bf*)(p);
  u.h[1] = *(const v8bf*)(p + 16);
  return u.v;
}

// Same fragment, but sourced from a row-major f32 matrix in global memory
// (used for B operands whose weight matrix is stored [out_col][k]).
__device__ __forceinline__ v16bf frag_from_global_f32(const float* W, int ld,
                                                      int r0, int k0, int lane) {
  int r  = r0 + (lane & 15);
  int kb = k0 + ((lane >> 4) << 3);
  const float* p = W + (size_t)r * ld + kb;
  v4f a0 = *(const v4f*)(p);
  v4f a1 = *(const v4f*)(p + 4);
  v4f a2 = *(const v4f*)(p + 16);
  v4f a3 = *(const v4f*)(p + 20);
  union { v16bf v; __bf16 e[16]; } u;
#pragma unroll
  for (int i = 0; i < 4; ++i) {
    u.e[i]      = (__bf16)a0[i];
    u.e[4 + i]  = (__bf16)a1[i];
    u.e[8 + i]  = (__bf16)a2[i];
    u.e[12 + i] = (__bf16)a3[i];
  }
  return u.v;
}

// ---------------------------------------------------------------------------
// Kernel 1: batched GRU over 40960 sequences (T=12, C=2, H=128).
// One wave handles a 16-sequence tile; recurrent matmul h @ Whh^T done with
// bf16 WMMA (f32 accumulate); xt @ Wih^T (K=2) folded into gate VALU math.
// blockDim = 128 (4 waves). Whh staged to LDS bf16 once per workgroup.
// ---------------------------------------------------------------------------
// LDS layout (dynamic):
//   whh_s : G3*HH bf16                    98304 B
//   wih_s : G3*CC f32                      3072 B
//   brz_s : 256 f32 (bih+bhh, r/z gates)   1024 B
//   bhn_s : 128 f32 (bhh, n gate)           512 B
//   bin_s : 128 f32 (bih, n gate)           512 B
//   hbuf  : 4 waves * 2 * 16*HH bf16      32768 B
//   xs    : 4 waves * 16*TT*CC f32         6144 B
constexpr size_t GRU_SMEM = 98304 + 3072 + 1024 + 512 + 512 + 32768 + 6144;

__global__ void __launch_bounds__(128)
gru_kernel(const float* __restrict__ pred, const float* __restrict__ Wih,
           const float* __restrict__ Whh, const float* __restrict__ bih,
           const float* __restrict__ bhh, float* __restrict__ hT) {
  extern __shared__ char smem[];
  __bf16* whh_s = (__bf16*)smem;
  float*  wih_s = (float*)(smem + 98304);
  float*  brz_s = (float*)(smem + 98304 + 3072);
  float*  bhn_s = brz_s + 256;
  float*  bin_s = bhn_s + 128;
  __bf16* hbuf  = (__bf16*)(bin_s + 128);
  float*  xs    = (float*)(hbuf + 4 * 2 * 16 * HH);

  const int tid = threadIdx.x;
  for (int i = tid; i < G3 * HH; i += 128) whh_s[i] = (__bf16)Whh[i];
  for (int i = tid; i < G3 * CC; i += 128) wih_s[i] = Wih[i];
  for (int i = tid; i < 256; i += 128) brz_s[i] = bih[i] + bhh[i];
  for (int i = tid; i < 128; i += 128) {
    bhn_s[i] = bhh[256 + i];
    bin_s[i] = bih[256 + i];
  }

  const int w    = tid >> 5;
  const int lane = tid & 31;
  const int s0   = (blockIdx.x * 4 + w) * 16;
  __bf16* hb = hbuf + w * 2 * 16 * HH;
  float*  xw = xs + w * 16 * TT * CC;

  // Stage x for this wave's 16 sequences: xw[row][t][c]
  for (int i = lane; i < 16 * TT * CC; i += 32) {
    int row = i / (TT * CC);
    int rem = i - row * (TT * CC);
    int t = rem >> 1, c = rem & 1;
    int s = s0 + row;
    int bn = s / MM, md = s - bn * MM;
    xw[i] = pred[(((size_t)bn * TT + t) * CC + c) * MM + md];
  }
  for (int i = lane; i < 16 * HH; i += 32) hb[i] = (__bf16)0.0f;
  __syncthreads();

  const int hl  = lane >> 4;
  const int nlo = lane & 15;
  float hreg[8][8];  // D-layout f32 hidden state: (m = v+8*hl, c = cb*16+nlo)
#pragma unroll
  for (int cb = 0; cb < 8; ++cb)
#pragma unroll
    for (int v = 0; v < 8; ++v) hreg[cb][v] = 0.0f;

  for (int t = 0; t < TT; ++t) {
    const __bf16* hcur = hb + (t & 1) * 16 * HH;
    __bf16*       hnxt = hb + ((t + 1) & 1) * 16 * HH;

    v16bf afr[4];
#pragma unroll
    for (int kk = 0; kk < 4; ++kk)
      afr[kk] = frag_from_lds(hcur, HH, 0, kk * 32, lane);

    float x0[8], x1[8];
#pragma unroll
    for (int v = 0; v < 8; ++v) {
      int m = v + 8 * hl;
      x0[v] = xw[m * (TT * CC) + t * CC + 0];
      x1[v] = xw[m * (TT * CC) + t * CC + 1];
    }

#pragma unroll
    for (int cb = 0; cb < 8; ++cb) {
      const int c = cb * 16 + nlo;
      const float wr0 = wih_s[c * 2 + 0],        wr1 = wih_s[c * 2 + 1];
      const float wz0 = wih_s[(128 + c) * 2 + 0], wz1 = wih_s[(128 + c) * 2 + 1];
      const float wn0 = wih_s[(256 + c) * 2 + 0], wn1 = wih_s[(256 + c) * 2 + 1];
      const float bin_v = bin_s[c];

      v8f accr  = splat8(brz_s[c]);
      v8f accz  = splat8(brz_s[128 + c]);
      v8f acchn = splat8(bhn_s[c]);
#pragma unroll
      for (int kk = 0; kk < 4; ++kk) {
        v16bf bR = frag_from_lds(whh_s, HH, cb * 16,       kk * 32, lane);
        v16bf bZ = frag_from_lds(whh_s, HH, 128 + cb * 16, kk * 32, lane);
        v16bf bN = frag_from_lds(whh_s, HH, 256 + cb * 16, kk * 32, lane);
        accr  = wmma_bf16(afr[kk], bR, accr);
        accz  = wmma_bf16(afr[kk], bZ, accz);
        acchn = wmma_bf16(afr[kk], bN, acchn);
      }
#pragma unroll
      for (int v = 0; v < 8; ++v) {
        float gr = accr[v] + wr0 * x0[v] + wr1 * x1[v];
        float gz = accz[v] + wz0 * x0[v] + wz1 * x1[v];
        float r = fast_sigmoid(gr);
        float z = fast_sigmoid(gz);
        float inn = bin_v + wn0 * x0[v] + wn1 * x1[v];
        float n = fast_tanh(inn + r * acchn[v]);
        float hnew = (1.0f - z) * n + z * hreg[cb][v];
        hreg[cb][v] = hnew;
        int m = v + 8 * hl;
        hnxt[m * HH + c] = (__bf16)hnew;
      }
    }
    __syncthreads();
  }

#pragma unroll
  for (int cb = 0; cb < 8; ++cb) {
    int c = cb * 16 + nlo;
#pragma unroll
    for (int v = 0; v < 8; ++v) {
      int m = v + 8 * hl;
      hT[(size_t)(s0 + m) * HH + c] = hreg[cb][v];
    }
  }
}

// ---------------------------------------------------------------------------
// Kernel 2: multi-head self-attention (4 heads, head_dim 32) + out proj +
// (masked) mean over sequence. One wave per group. SPAD = padded seq (mult 16).
// ---------------------------------------------------------------------------
template <int SPAD>
constexpr size_t attn_smem() {
  return (size_t)SPAD * 128 * 4      // attf (f32 attention output)
       + (size_t)SPAD * SPAD * 4     // sc   (f32 scores)
       + 4u * SPAD * 128 * 2         // xbf, qbf, kbf, obf
       + (size_t)128 * SPAD * 2      // vtb (v transposed)
       + (size_t)SPAD * SPAD * 2;    // pbf (probs bf16)
}

template <int SPAD>
__global__ void __launch_bounds__(32)
attn_kernel(const float* __restrict__ X, const float* __restrict__ inw,
            const float* __restrict__ inb, const float* __restrict__ outw,
            const float* __restrict__ outb, const float* __restrict__ mask,
            float* __restrict__ agg, int S) {
  extern __shared__ char smem[];
  float*  attf = (float*)smem;
  float*  sc   = attf + SPAD * 128;
  __bf16* xbf  = (__bf16*)(sc + SPAD * SPAD);
  __bf16* qbf  = xbf + SPAD * 128;
  __bf16* kbf  = qbf + SPAD * 128;
  __bf16* vtb  = kbf + SPAD * 128;   // [dim][key]
  __bf16* pbf  = vtb + 128 * SPAD;   // [row][key]
  __bf16* obf  = pbf + SPAD * SPAD;  // [row][dim]

  const int lane = threadIdx.x;
  const int g    = blockIdx.x;
  const int hl   = lane >> 4;
  const int nlo  = lane & 15;
  const float rs = 0.17677669529663687f;  // 1/sqrt(32)

  for (int i = lane; i < SPAD * 128; i += 32) {
    int r = i >> 7;
    xbf[i] = (r < S) ? (__bf16)X[(size_t)(g * S + r) * 128 + (i & 127)]
                     : (__bf16)0.0f;
  }
  __syncthreads();

  // qkv = X @ inw^T + inb
  for (int mt = 0; mt < SPAD / 16; ++mt) {
    for (int nt = 0; nt < 24; ++nt) {
      v8f acc = splat8(inb[nt * 16 + nlo]);
#pragma unroll
      for (int kk = 0; kk < 4; ++kk) {
        v16bf a = frag_from_lds(xbf, 128, mt * 16, kk * 32, lane);
        v16bf b = frag_from_global_f32(inw, 128, nt * 16, kk * 32, lane);
        acc = wmma_bf16(a, b, acc);
      }
      int col = nt * 16 + nlo;
      int gate = col >> 7;  // uniform per nt
      int cc = col & 127;
#pragma unroll
      for (int v = 0; v < 8; ++v) {
        int m = mt * 16 + v + 8 * hl;
        float val = acc[v];
        if (gate == 0)      qbf[m * 128 + cc] = (__bf16)(val * rs);
        else if (gate == 1) kbf[m * 128 + cc] = (__bf16)val;
        else                vtb[cc * SPAD + m] = (__bf16)val;
      }
    }
  }
  __syncthreads();

  for (int hd = 0; hd < 4; ++hd) {
    // scores = q @ k^T  (K = 32 -> one WMMA per tile)
#pragma unroll
    for (int mt = 0; mt < SPAD / 16; ++mt)
#pragma unroll
      for (int nt = 0; nt < SPAD / 16; ++nt) {
        v8f acc = splat8(0.0f);
        v16bf a = frag_from_lds(qbf, 128, mt * 16, hd * 32, lane);
        v16bf b = frag_from_lds(kbf, 128, nt * 16, hd * 32, lane);
        acc = wmma_bf16(a, b, acc);
#pragma unroll
        for (int v = 0; v < 8; ++v)
          sc[(mt * 16 + v + 8 * hl) * SPAD + nt * 16 + nlo] = acc[v];
      }
    __syncthreads();

    // row softmax with key-padding mask
    for (int r = lane; r < SPAD; r += 32) {
      if (r < S) {
        float mx = -1e30f;
        for (int j = 0; j < S; ++j) {
          float sv = sc[r * SPAD + j];
          if (mask && mask[g * S + j] == 0.0f) sv = -1e9f;
          sc[r * SPAD + j] = sv;
          mx = fmaxf(mx, sv);
        }
        float sum = 0.0f;
        for (int j = 0; j < S; ++j) {
          float e = __expf(sc[r * SPAD + j] - mx);
          sc[r * SPAD + j] = e;
          sum += e;
        }
        float inv = fast_rcp(sum);
        for (int j = 0; j < S; ++j) pbf[r * SPAD + j] = (__bf16)(sc[r * SPAD + j] * inv);
        for (int j = S; j < SPAD; ++j) pbf[r * SPAD + j] = (__bf16)0.0f;
      } else {
        for (int j = 0; j < SPAD; ++j) pbf[r * SPAD + j] = (__bf16)0.0f;
      }
    }
    __syncthreads();

    // o = probs @ v  (this head's 32 output dims)
#pragma unroll
    for (int mt = 0; mt < SPAD / 16; ++mt)
#pragma unroll
      for (int nt = 0; nt < 2; ++nt) {
        v8f acc = splat8(0.0f);
#pragma unroll
        for (int kk = 0; kk < SPAD / 32; ++kk) {
          v16bf a = frag_from_lds(pbf, SPAD, mt * 16, kk * 32, lane);
          v16bf b = frag_from_lds(vtb, SPAD, hd * 32 + nt * 16, kk * 32, lane);
          acc = wmma_bf16(a, b, acc);
        }
#pragma unroll
        for (int v = 0; v < 8; ++v)
          obf[(mt * 16 + v + 8 * hl) * 128 + hd * 32 + nt * 16 + nlo] =
              (__bf16)acc[v];
      }
    __syncthreads();
  }

  // out projection: att = o @ outw^T + outb
  for (int mt = 0; mt < SPAD / 16; ++mt) {
    for (int nt = 0; nt < 8; ++nt) {
      v8f acc = splat8(outb[nt * 16 + nlo]);
#pragma unroll
      for (int kk = 0; kk < 4; ++kk) {
        v16bf a = frag_from_lds(obf, 128, mt * 16, kk * 32, lane);
        v16bf b = frag_from_global_f32(outw, 128, nt * 16, kk * 32, lane);
        acc = wmma_bf16(a, b, acc);
      }
#pragma unroll
      for (int v = 0; v < 8; ++v)
        attf[(mt * 16 + v + 8 * hl) * 128 + nt * 16 + nlo] = acc[v];
    }
  }
  __syncthreads();

  // (masked) mean over rows
  float nv;
  if (mask) {
    float cnt = 0.0f;
    for (int j = 0; j < S; ++j) cnt += (mask[g * S + j] != 0.0f) ? 1.0f : 0.0f;
    nv = fmaxf(cnt, 1.0f);
  } else {
    nv = (float)S;
  }
  const float invnv = fast_rcp(nv);
  for (int i = lane; i < 128; i += 32) {
    float acc = 0.0f;
    for (int r = 0; r < S; ++r) {
      float wgt = mask ? ((mask[g * S + r] != 0.0f) ? 1.0f : 0.0f) : 1.0f;
      acc += attf[r * 128 + i] * wgt;
    }
    agg[(size_t)g * 128 + i] = acc * invnv;
  }
}

// ---------------------------------------------------------------------------
// Kernel 3: Y[row] = LayerNorm(relu(X[row] @ W^T + b)) * (mask ? mask[row] : 1)
// blockDim == out_dim (64 or 128); one row per block.
// ---------------------------------------------------------------------------
__global__ void __launch_bounds__(128)
mlp_ln_kernel(const float* __restrict__ X, const float* __restrict__ W,
              const float* __restrict__ bias, const float* __restrict__ lnw,
              const float* __restrict__ lnb, const float* __restrict__ mask,
              float* __restrict__ Y, int in_dim, int out_dim) {
  __shared__ float xrow[128];
  __shared__ float red[128];
  const int row = blockIdx.x;
  const int j   = threadIdx.x;

  for (int i = j; i < in_dim; i += blockDim.x) xrow[i] = X[(size_t)row * in_dim + i];
  __syncthreads();

  float acc = bias[j];
  for (int k = 0; k < in_dim; ++k) acc += xrow[k] * W[(size_t)j * in_dim + k];
  acc = fmaxf(acc, 0.0f);

  const float inv_dim = fast_rcp((float)out_dim);

  red[j] = acc;
  __syncthreads();
  for (int s = blockDim.x >> 1; s > 0; s >>= 1) {
    if (j < s) red[j] += red[j + s];
    __syncthreads();
  }
  float mean = red[0] * inv_dim;
  __syncthreads();

  float d = acc - mean;
  red[j] = d * d;
  __syncthreads();
  for (int s = blockDim.x >> 1; s > 0; s >>= 1) {
    if (j < s) red[j] += red[j + s];
    __syncthreads();
  }
  float var = red[0] * inv_dim;

  float y = d * __frsqrt_rn(var + 1e-5f) * lnw[j] + lnb[j];
  if (mask) y *= mask[row];
  Y[(size_t)row * out_dim + j] = y;
}

// ---------------------------------------------------------------------------
// Host launch
// ---------------------------------------------------------------------------
extern "C" void kernel_launch(void* const* d_in, const int* in_sizes, int n_in,
                              void* d_out, int out_size, void* d_ws,
                              size_t ws_size, hipStream_t stream) {
  (void)in_sizes; (void)n_in; (void)out_size; (void)ws_size;

  const float* pred    = (const float*)d_in[0];
  const float* pmask   = (const float*)d_in[1];
  const float* gWih    = (const float*)d_in[2];
  const float* gWhh    = (const float*)d_in[3];
  const float* gbih    = (const float*)d_in[4];
  const float* gbhh    = (const float*)d_in[5];
  const float* minw    = (const float*)d_in[6];
  const float* minb    = (const float*)d_in[7];
  const float* moutw   = (const float*)d_in[8];
  const float* moutb   = (const float*)d_in[9];
  const float* maggw   = (const float*)d_in[10];
  const float* maggb   = (const float*)d_in[11];
  const float* magglnw = (const float*)d_in[12];
  const float* magglnb = (const float*)d_in[13];
  const float* pinw    = (const float*)d_in[14];
  const float* pinb    = (const float*)d_in[15];
  const float* poutw   = (const float*)d_in[16];
  const float* poutb   = (const float*)d_in[17];
  const float* paggw   = (const float*)d_in[18];
  const float* paggb   = (const float*)d_in[19];
  const float* pagglnw = (const float*)d_in[20];
  const float* pagglnb = (const float*)d_in[21];
  const float* outw    = (const float*)d_in[22];
  const float* outb    = (const float*)d_in[23];
  const float* outlnw  = (const float*)d_in[24];
  const float* outlnb  = (const float*)d_in[25];

  float* ws    = (float*)d_ws;
  float* hT    = ws;                               // 40960*128
  float* amean = hT + (size_t)NSEQ * HH;           // 2048*128
  float* pedf  = amean + (size_t)NBN * HH;         // 2048*128
  float* pag1  = pedf + (size_t)NBN * HH;          // 32*128
  float* pag2  = pag1 + BB * HH;                   // 32*128

  // 1) GRU over all (b,n,mode) sequences -> hT
  gru_kernel<<<NSEQ / 64, 128, GRU_SMEM, stream>>>(pred, gWih, gWhh, gbih,
                                                   gbhh, hT);

  // 2) mode self-attention (S=20, pad 32) + mean pool -> amean
  attn_kernel<32><<<NBN, 32, attn_smem<32>(), stream>>>(
      hT, minw, minb, moutw, moutb, nullptr, amean, MM);

  // 3) magg: linear+relu+LN, then * pedestrian_mask -> pedf
  mlp_ln_kernel<<<NBN, 128, 0, stream>>>(amean, maggw, maggb, magglnw,
                                         magglnb, pmask, pedf, HH, HH);

  // 4) pedestrian attention (S=64) with key-padding mask + masked mean -> pag1
  attn_kernel<64><<<BB, 32, attn_smem<64>(), stream>>>(
      pedf, pinw, pinb, poutw, poutb, pmask, pag1, NN);

  // 5) pagg: linear+relu+LN -> pag2
  mlp_ln_kernel<<<BB, 128, 0, stream>>>(pag1, paggw, paggb, pagglnw, pagglnb,
                                        nullptr, pag2, HH, HH);

  // 6) out: linear(H->F)+relu+LN -> d_out (32 x 64)
  mlp_ln_kernel<<<BB, FF, 0, stream>>>(pag2, outw, outb, outlnw, outlnb,
                                       nullptr, (float*)d_out, HH, FF);
}
